// SinkhornLossPOT_48576080118112
// MI455X (gfx1250) — matrix operand order repaired and checked by reading, not compiled, not run on hardware
//
#include <hip/hip_runtime.h>
#include <hip/hip_bf16.h>

typedef __attribute__((ext_vector_type(16))) _Float16 v16h;
typedef __attribute__((ext_vector_type(8)))  _Float16 v8h;
typedef __attribute__((ext_vector_type(8)))  float    v8f;
typedef __attribute__((ext_vector_type(4)))  int      v4i;

#define SK_N 8192
#define SK_D 256
#define SK_NKC 8            // 256 / 32 K-chunks per tile
#define SK_MT 32            // rows staged per pipeline step (2 sub-tiles of 16)
#define SK_INVREG 20.0f     // 1/0.05
#define SK_LOGMARG (-9.0109131f) // -log(8192)

// ---- CDNA5 async global->LDS staging (ASYNCcnt path), with sync fallback ----
#if defined(__has_builtin)
# if __has_builtin(__builtin_amdgcn_global_load_async_to_lds_b128)
#  define SK_ASYNC 1
# endif
#endif
#ifndef SK_ASYNC
# define SK_ASYNC 0
#endif

#if SK_ASYNC
typedef __attribute__((address_space(1))) v4i sk_gv4i; // global 128-bit chunk
typedef __attribute__((address_space(3))) v4i sk_lv4i; // LDS 128-bit chunk
# if __has_builtin(__builtin_amdgcn_s_wait_asynccnt)
#  define SK_WAIT_ASYNC() __builtin_amdgcn_s_wait_asynccnt(0)
# else
#  define SK_WAIT_ASYNC() asm volatile("s_wait_asynccnt 0x0" ::: "memory")
# endif
#else
# define SK_WAIT_ASYNC() ((void)0)
#endif

// Each of 256 threads moves 64 contiguous bytes of a 16 KB (SK_MT x 256 f16) tile.
__device__ __forceinline__ void sk_stage(const _Float16* gsrc, v16h* lbuf, int tid) {
#if SK_ASYNC
  char* g = (char*)gsrc + tid * 64;
  char* l = (char*)lbuf + tid * 64;
#pragma unroll
  for (int q = 0; q < 4; ++q)
    __builtin_amdgcn_global_load_async_to_lds_b128(
        (sk_gv4i*)(g + 16 * q), (sk_lv4i*)(l + 16 * q), 0, 0);
#else
  const v16h* g = ((const v16h*)gsrc) + tid * 2;
  lbuf[tid * 2]     = g[0];
  lbuf[tid * 2 + 1] = g[1];
#endif
}

// Build a 16x32 f16 A-fragment from a row-major 16x256 tile in `base`.
// Contiguous K order (matches B fragments): lanes 0-15 hold K = k0+[0..15];
// lanes 16-31 hold K = k0+16+[0..15]; row M = lane&15.
__device__ __forceinline__ v16h sk_a_frag(const _Float16* base, int r, int k0, int hi) {
  return *(const v16h*)(base + r * SK_D + k0 + hi * 16);
}

// Precompute f16 copies of x,y, squared row norms, and zero potentials.
__global__ __launch_bounds__(256) void sinkhorn_prep(
    const float* __restrict__ x, const float* __restrict__ y,
    _Float16* __restrict__ xh, _Float16* __restrict__ yh,
    float* __restrict__ x2, float* __restrict__ y2,
    float* __restrict__ u, float* __restrict__ v) {
  __shared__ float redx[256];
  __shared__ float redy[256];
  const int row = blockIdx.x;
  const int t = threadIdx.x;
  const float xv = x[(size_t)row * SK_D + t];
  const float yv = y[(size_t)row * SK_D + t];
  xh[(size_t)row * SK_D + t] = (_Float16)xv;
  yh[(size_t)row * SK_D + t] = (_Float16)yv;
  redx[t] = xv * xv;
  redy[t] = yv * yv;
  __syncthreads();
  for (int s = 128; s > 0; s >>= 1) {
    if (t < s) { redx[t] += redx[t + s]; redy[t] += redy[t + s]; }
    __syncthreads();
  }
  if (t == 0) {
    x2[row] = redx[0];
    y2[row] = redy[0];
    u[row] = 0.0f;
    v[row] = 0.0f;
  }
}

// One Sinkhorn half-step in log domain:
//   pout[n] = -log(N) - logsumexp_m( min(40*S[m][n] - 20*sSq[m] - 20*oSq[n], 0) + pin[m] )
// S[m][n] = dot(Sh_row_m, Oh_row_n) via WMMA f16 GEMM, f32 accum; staged side
// double-buffered through LDS with async global->LDS loads.
__global__ __launch_bounds__(256) void sinkhorn_half_step(
    const _Float16* __restrict__ Sh, const _Float16* __restrict__ Oh,
    const float* __restrict__ sSq, const float* __restrict__ oSq,
    const float* __restrict__ pin, float* __restrict__ pout) {
  __shared__ v16h stage[2][SK_MT * 16]; // 2 x (SK_MT rows x 256 halfs) = 32 KB

  const int tid  = threadIdx.x;
  const int wave = tid >> 5;
  const int lane = tid & 31;
  const int l15  = lane & 15;
  const int hi   = lane >> 4;

  const int j = (blockIdx.x * 8 + wave) * 16 + l15; // owned column

  // Preload owned-side B fragments (16 rows of Oh as the B matrix, K=256).
  v16h bfrag[SK_NKC];
  const _Float16* orow = Oh + (size_t)j * SK_D;
#pragma unroll
  for (int kk = 0; kk < SK_NKC; ++kk)
    bfrag[kk] = *(const v16h*)(orow + kk * 32 + hi * 16);

  const float oq = -SK_INVREG * oSq[j];

  float mrun = -__builtin_inff();
  float srun = 0.0f;

  // Pipeline prologue: stage tile 0.
  sk_stage(Sh, stage[0], tid);

  int buf = 0;
  for (int m0 = 0; m0 < SK_N; m0 += SK_MT) {
    SK_WAIT_ASYNC();      // this wave's async loads into stage[buf] landed
    __syncthreads();      // => every wave's loads landed; prev compute done

    if (m0 + SK_MT < SK_N)
      sk_stage(Sh + (size_t)(m0 + SK_MT) * SK_D, stage[buf ^ 1], tid);

    const _Float16* sm = (const _Float16*)stage[buf];

#pragma unroll
    for (int sub = 0; sub < SK_MT / 16; ++sub) {
      const _Float16* smt = sm + sub * 16 * SK_D;
      v8f c = {};
#pragma unroll
      for (int kk = 0; kk < SK_NKC; ++kk) {
        v16h a = sk_a_frag(smt, l15, kk * 32, hi);
        c = __builtin_amdgcn_wmma_f32_16x16x32_f16(false, a, false, bfrag[kk],
                                                   (short)0, c, false, false);
      }

      // t = min(40*s - 20*x2 - 20*y2, 0) + pin[m]; online logsumexp over m.
      const int mbase = m0 + sub * 16 + hi * 8;
      float tv[8];
      float tmax = -3.4e38f;
#pragma unroll
      for (int r = 0; r < 8; ++r) {
        float q = 40.0f * c[r] - SK_INVREG * sSq[mbase + r] + oq;
        float t = fminf(q, 0.0f) + pin[mbase + r];
        tv[r] = t;
        tmax = fmaxf(tmax, t);
      }
      tmax = fmaxf(tmax, __shfl_xor(tmax, 16, 32)); // merge M halves per column
      const float nm = fmaxf(mrun, tmax);
      float p = 0.0f;
#pragma unroll
      for (int r = 0; r < 8; ++r) p += __expf(tv[r] - nm);
      p += __shfl_xor(p, 16, 32);
      srun = srun * __expf(mrun - nm) + p;
      mrun = nm;
    }
    buf ^= 1;
  }

  if (lane < 16) pout[j] = SK_LOGMARG - (mrun + __logf(srun));
}

__global__ void sinkhorn_zero(float* __restrict__ out) { out[0] = 0.0f; }

// loss = sum_{i,j} exp(-20*d2 + u[i] + v[j]) * d2,  d2 = max(x2+y2-2*s, 0)
__global__ __launch_bounds__(256) void sinkhorn_loss(
    const _Float16* __restrict__ xh, const _Float16* __restrict__ yh,
    const float* __restrict__ x2, const float* __restrict__ y2,
    const float* __restrict__ u, const float* __restrict__ v,
    float* __restrict__ out) {
  __shared__ v16h stage[256];
  const _Float16* sm = (const _Float16*)stage;

  const int tid  = threadIdx.x;
  const int wave = tid >> 5;
  const int lane = tid & 31;
  const int l15  = lane & 15;
  const int hi   = lane >> 4;

  const int j = (blockIdx.x * 8 + wave) * 16 + l15;

  v16h bfrag[SK_NKC];
  const _Float16* orow = yh + (size_t)j * SK_D;
#pragma unroll
  for (int kk = 0; kk < SK_NKC; ++kk)
    bfrag[kk] = *(const v16h*)(orow + kk * 32 + hi * 16);

  const float y2j = y2[j];
  const float vj  = v[j];

  float acc = 0.0f;

  for (int m0 = 0; m0 < SK_N; m0 += 16) {
    __syncthreads();
    stage[tid] = *(((const v16h*)(xh + (size_t)m0 * SK_D)) + tid);
    __syncthreads();

    v8f c = {};
#pragma unroll
    for (int kk = 0; kk < SK_NKC; ++kk) {
      v16h a = sk_a_frag(sm, l15, kk * 32, hi);
      c = __builtin_amdgcn_wmma_f32_16x16x32_f16(false, a, false, bfrag[kk],
                                                 (short)0, c, false, false);
    }

    const int mbase = m0 + hi * 8;
#pragma unroll
    for (int r = 0; r < 8; ++r) {
      float d2 = fmaxf(x2[mbase + r] + y2j - 2.0f * c[r], 0.0f);
      acc += __expf(-SK_INVREG * d2 + u[mbase + r] + vj) * d2;
    }
  }

#pragma unroll
  for (int off = 16; off >= 1; off >>= 1) acc += __shfl_xor(acc, off, 32);
  if (lane == 0) atomicAdd(out, acc);
}

extern "C" void kernel_launch(void* const* d_in, const int* in_sizes, int n_in,
                              void* d_out, int out_size, void* d_ws, size_t ws_size,
                              hipStream_t stream) {
  const float* x = (const float*)d_in[0];
  const float* y = (const float*)d_in[1];
  float* out = (float*)d_out;

  _Float16* xh = (_Float16*)d_ws;
  _Float16* yh = xh + (size_t)SK_N * SK_D;
  float* x2 = (float*)(yh + (size_t)SK_N * SK_D);
  float* y2 = x2 + SK_N;
  float* u  = y2 + SK_N;
  float* v  = u + SK_N;

  sinkhorn_prep<<<SK_N, 256, 0, stream>>>(x, y, xh, yh, x2, y2, u, v);

  for (int it = 0; it < 100; ++it) {
    // v = logb - lse_i(Mr + u[i])   (staged = x, owned = y)
    sinkhorn_half_step<<<64, 256, 0, stream>>>(xh, yh, x2, y2, u, v);
    // u = loga - lse_j(Mr + v[j])   (staged = y, owned = x)
    sinkhorn_half_step<<<64, 256, 0, stream>>>(yh, xh, y2, x2, v, u);
  }

  sinkhorn_zero<<<1, 1, 0, stream>>>(out);
  sinkhorn_loss<<<64, 256, 0, stream>>>(xh, yh, x2, y2, u, v, out);
}
